// MultiHead2_43885975830734
// MI455X (gfx1250) — compile-verified, matrix operand order checked by reading, not run on hardware
//
#include <hip/hip_runtime.h>
#include <hip/hip_bf16.h>

// ---- problem constants (from reference) ----
#define B_  4
#define I_  2048
#define E_  768
#define HS_ 64
#define H_  12

typedef __attribute__((ext_vector_type(16))) __bf16         v16bf;
typedef __attribute__((ext_vector_type(8)))  float          v8f;
typedef __attribute__((ext_vector_type(8)))  unsigned short ushort8;
typedef __attribute__((ext_vector_type(4)))  float          f32x4;

union BF16V {
    unsigned short u[16];
    ushort8        h[2];
    v16bf          v;
};

static __device__ __forceinline__ unsigned short f2bf(float f) {
    unsigned int u = __float_as_uint(f);
    u += 0x7FFFu + ((u >> 16) & 1u);   // round-to-nearest-even
    return (unsigned short)(u >> 16);
}

// ---------------- kernel 0: weight transpose fp32[E,HS] -> bf16[HS,E] x3 ----
__global__ void MultiHead2_wtrans(const float* __restrict__ Wq,
                                  const float* __restrict__ Wk,
                                  const float* __restrict__ Wv,
                                  unsigned short* __restrict__ WT) {
    int idx = blockIdx.x * blockDim.x + threadIdx.x;
    if (idx >= 3 * HS_ * E_) return;
    int mat = idx / (HS_ * E_);
    int rem = idx - mat * (HS_ * E_);
    int n = rem / E_;          // column of W -> row of WT
    int k = rem - n * E_;      // row of W    -> col of WT
    const float* W = (mat == 0) ? Wq : (mat == 1) ? Wk : Wv;
    WT[idx] = f2bf(W[k * HS_ + n]);
}

// ---------------- kernel 1: fused Q/K/V projection via bf16 WMMA -------------
// One wave computes the 16x64 tiles of Q, K AND V for its row tile: the x tile
// is loaded and converted to bf16 exactly once (x is the dominant HBM traffic).
// 12 accumulators (3 mats x 4 N-tiles), 12 WMMAs per K-step, 24 K-steps.
__global__ __launch_bounds__(128) void MultiHead2_qkv(
        const float* __restrict__ x, const unsigned short* __restrict__ WT,
        unsigned short* __restrict__ Qs, unsigned short* __restrict__ Kb,
        unsigned short* __restrict__ Vt) {
    const int lane = threadIdx.x & 31;
    const int warp = threadIdx.x >> 5;
    const int l16  = lane & 15;
    const int hi   = lane >> 4;

    int tile = blockIdx.x * 4 + warp;     // 0..511
    int b    = tile >> 7;
    int r0   = (tile & 127) << 4;

    const float* xrow = x + (size_t)(b * I_ + r0 + l16) * E_;

    v8f acc[3][4];
    #pragma unroll
    for (int m = 0; m < 3; ++m)
        #pragma unroll
        for (int t = 0; t < 4; ++t) acc[m][t] = (v8f)(0.0f);

    for (int kk = 0; kk < E_; kk += 32) {
        const int s0 = kk + hi * 8;       // A/B element->K mapping (ISA 7.12.2)
        const int s1 = kk + 16 + hi * 8;
        // A operand: x tile row, convert f32 -> bf16 (once for all 3 matrices)
        BF16V a;
        f32x4 x0 = *(const f32x4*)(xrow + s0);
        f32x4 x1 = *(const f32x4*)(xrow + s0 + 4);
        f32x4 x2 = *(const f32x4*)(xrow + s1);
        f32x4 x3 = *(const f32x4*)(xrow + s1 + 4);
        #pragma unroll
        for (int e = 0; e < 4; ++e) {
            a.u[e]      = f2bf(x0[e]);
            a.u[e + 4]  = f2bf(x1[e]);
            a.u[e + 8]  = f2bf(x2[e]);
            a.u[e + 12] = f2bf(x3[e]);
        }
        #pragma unroll
        for (int m = 0; m < 3; ++m) {
            const unsigned short* wmat = WT + (size_t)m * (HS_ * E_);
            #pragma unroll
            for (int t = 0; t < 4; ++t) {
                const unsigned short* wrow = wmat + (size_t)(t * 16 + l16) * E_;
                BF16V bv;
                bv.h[0] = *(const ushort8*)(wrow + s0);
                bv.h[1] = *(const ushort8*)(wrow + s1);
                acc[m][t] = __builtin_amdgcn_wmma_f32_16x16x32_bf16(
                    false, a.v, false, bv.v, (short)0, acc[m][t], false, false);
            }
        }
    }

    const float qscale = 0.125f;          // 1/sqrt(64)
    #pragma unroll
    for (int t = 0; t < 4; ++t) {
        #pragma unroll
        for (int r = 0; r < 8; ++r) {
            int row = r0 + r + 8 * hi;    // C layout: VGPR r -> M=r (+8 hi half)
            int col = t * 16 + l16;
            Qs[(size_t)(b * I_ + row) * HS_ + col] = f2bf(acc[0][t][r] * qscale);
            Kb[(size_t)(b * I_ + row) * HS_ + col] = f2bf(acc[1][t][r]);
            Vt[((size_t)b * HS_ + col) * I_ + row] = f2bf(acc[2][t][r]);
        }
    }
}

// ---------------- kernel 2: causal flash attention, bf16 WMMA ----------------
// One wave per 16-query tile; 32 keys per step (scores: 2 tiles x K=64,
// PV: K=32 exactly one WMMA per 16-col output tile).
__global__ __launch_bounds__(128) void MultiHead2_attn(
        const unsigned short* __restrict__ Qs,
        const unsigned short* __restrict__ Kb,
        const unsigned short* __restrict__ Vt,
        float* __restrict__ out) {
    __shared__ unsigned short pstage[4 * 16 * 32];   // per-wave 16x32 bf16 P tile

    const int lane = threadIdx.x & 31;
    const int warp = threadIdx.x >> 5;
    const int l16  = lane & 15;
    const int hi   = lane >> 4;

    int job = blockIdx.x * 4 + warp;      // 0..511
    int b   = job >> 7;
    int q0  = (job & 127) << 4;

    // Preload Q A-operand (two K=32 chunks covering HS=64)
    const unsigned short* qrow = Qs + (size_t)(b * I_ + q0 + l16) * HS_;
    BF16V aq[2];
    #pragma unroll
    for (int c = 0; c < 2; ++c) {
        int s0 = c * 32 + hi * 8;
        aq[c].h[0] = *(const ushort8*)(qrow + s0);
        aq[c].h[1] = *(const ushort8*)(qrow + s0 + 16);
    }

    v8f o[4];
    #pragma unroll
    for (int t = 0; t < 4; ++t) o[t] = (v8f)(0.0f);
    float rm[8], rl[8];
    #pragma unroll
    for (int r = 0; r < 8; ++r) { rm[r] = -3.0e38f; rl[r] = 0.0f; }

    unsigned short*       myp   = pstage + warp * (16 * 32);
    const unsigned short* kbase = Kb + (size_t)b * I_ * HS_;
    const unsigned short* vbase = Vt + (size_t)b * HS_ * I_;

    const int jend = q0 + 16;             // cols needed: 0..q0+15 (causal)
    for (int j = 0; j < jend; j += 32) {
        // prefetch next key/value block into cache hierarchy
        __builtin_prefetch(kbase + (size_t)(j + 32 + l16) * HS_, 0, 3);
        __builtin_prefetch(vbase + (size_t)l16 * I_ + j + 32, 0, 3);

        // ---- scores: S = Q * K^T for 32 key columns (two 16-wide tiles) ----
        const unsigned short* krow0 = kbase + (size_t)(j + l16) * HS_;
        const unsigned short* krow1 = kbase + (size_t)(j + 16 + l16) * HS_;
        v8f s0 = (v8f)(0.0f), s1 = (v8f)(0.0f);
        #pragma unroll
        for (int c = 0; c < 2; ++c) {
            int off = c * 32 + hi * 8;
            BF16V bk0, bk1;
            bk0.h[0] = *(const ushort8*)(krow0 + off);
            bk0.h[1] = *(const ushort8*)(krow0 + off + 16);
            bk1.h[0] = *(const ushort8*)(krow1 + off);
            bk1.h[1] = *(const ushort8*)(krow1 + off + 16);
            s0 = __builtin_amdgcn_wmma_f32_16x16x32_bf16(
                false, aq[c].v, false, bk0.v, (short)0, s0, false, false);
            s1 = __builtin_amdgcn_wmma_f32_16x16x32_bf16(
                false, aq[c].v, false, bk1.v, (short)0, s1, false, false);
        }

        // ---- causal mask + online softmax (row reduce over 16-lane half) ----
        const int n0 = j + l16, n1 = j + 16 + l16;
        float p0[8], p1[8];
        #pragma unroll
        for (int r = 0; r < 8; ++r) {
            int m = q0 + r + 8 * hi;
            p0[r] = (n0 <= m) ? s0[r] : -1.0e30f;
            p1[r] = (n1 <= m) ? s1[r] : -1.0e30f;
        }
        #pragma unroll
        for (int r = 0; r < 8; ++r) {
            float t = fmaxf(p0[r], p1[r]);
            t = fmaxf(t, __shfl_xor(t, 1, 32));
            t = fmaxf(t, __shfl_xor(t, 2, 32));
            t = fmaxf(t, __shfl_xor(t, 4, 32));
            t = fmaxf(t, __shfl_xor(t, 8, 32));
            float mnew  = fmaxf(rm[r], t);
            float alpha = __expf(rm[r] - mnew);
            rm[r] = mnew;
            float e0 = __expf(p0[r] - mnew);
            float e1 = __expf(p1[r] - mnew);
            p0[r] = e0; p1[r] = e1;
            float s = e0 + e1;
            s += __shfl_xor(s, 1, 32);
            s += __shfl_xor(s, 2, 32);
            s += __shfl_xor(s, 4, 32);
            s += __shfl_xor(s, 8, 32);
            rl[r] = rl[r] * alpha + s;
            #pragma unroll
            for (int t4 = 0; t4 < 4; ++t4) o[t4][r] *= alpha;
        }

        // ---- stage P (C-layout -> row-major 16x32 bf16 in LDS) ----
        #pragma unroll
        for (int r = 0; r < 8; ++r) {
            int row = r + 8 * hi;
            myp[row * 32 + l16]      = f2bf(p0[r]);
            myp[row * 32 + 16 + l16] = f2bf(p1[r]);
        }
        // DS ops from one wave execute in order -> safe to reload as A operand
        BF16V ap;
        ap.h[0] = *(const ushort8*)(myp + l16 * 32 + hi * 8);
        ap.h[1] = *(const ushort8*)(myp + l16 * 32 + 16 + hi * 8);

        // ---- O += P * V (K=32, one WMMA per 16-col output tile) ----
        #pragma unroll
        for (int t = 0; t < 4; ++t) {
            const unsigned short* vrow = vbase + (size_t)(t * 16 + l16) * I_ + j;
            BF16V bv;
            bv.h[0] = *(const ushort8*)(vrow + hi * 8);
            bv.h[1] = *(const ushort8*)(vrow + 16 + hi * 8);
            o[t] = __builtin_amdgcn_wmma_f32_16x16x32_bf16(
                false, ap.v, false, bv.v, (short)0, o[t], false, false);
        }
    }

    // ---- normalize and write, tiled x12 across E ----
    #pragma unroll
    for (int r = 0; r < 8; ++r) {
        float inv = 1.0f / rl[r];
        int   row = q0 + r + 8 * hi;
        float* orow = out + (size_t)(b * I_ + row) * E_;
        #pragma unroll
        for (int t = 0; t < 4; ++t) {
            float val = o[t][r] * inv;
            int   col = t * 16 + l16;
            #pragma unroll
            for (int h = 0; h < H_; ++h) orow[h * HS_ + col] = val;
        }
    }
}

extern "C" void kernel_launch(void* const* d_in, const int* in_sizes, int n_in,
                              void* d_out, int out_size, void* d_ws, size_t ws_size,
                              hipStream_t stream) {
    (void)in_sizes; (void)n_in; (void)out_size; (void)ws_size;
    const float* x  = (const float*)d_in[0];
    const float* Wq = (const float*)d_in[1];
    const float* Wk = (const float*)d_in[2];
    const float* Wv = (const float*)d_in[3];
    float* out = (float*)d_out;

    // workspace layout (bf16/ushort): Qs | Kb | Vt | WT   (~3.3 MB total)
    unsigned short* Qs = (unsigned short*)d_ws;             // [B,I,HS]
    unsigned short* Kb = Qs + (size_t)B_ * I_ * HS_;        // [B,I,HS]
    unsigned short* Vt = Kb + (size_t)B_ * I_ * HS_;        // [B,HS,I]
    unsigned short* WT = Vt + (size_t)B_ * HS_ * I_;        // [3,HS,E]

    MultiHead2_wtrans<<<(3 * HS_ * E_ + 255) / 256, 256, 0, stream>>>(Wq, Wk, Wv, WT);
    MultiHead2_qkv <<<128, 128, 0, stream>>>(x, WT, Qs, Kb, Vt);
    MultiHead2_attn<<<128, 128, 0, stream>>>(Qs, Kb, Vt, out);
}
